// Block_74689481277452
// MI455X (gfx1250) — matrix operand order repaired and checked by reading, not compiled, not run on hardware
//
#include <hip/hip_runtime.h>

typedef __bf16 bf16;
typedef __attribute__((ext_vector_type(16))) __bf16 v16bf;
typedef __attribute__((ext_vector_type(8)))  float  v8f;

// problem constants
#define Bq    4
#define Tt    1024
#define TIMGc 512
#define Cc    1024
#define Hh    16
#define HDd   64
#define Ee    8
#define FFf   4096
#define Ntok  (Bq*Tt)
#define NimgT (Bq*TIMGc)
#define EPSc  1e-5f

// GEMM tiling
#define BM  128
#define BN  128
#define KBc 32
#define PAD 8

// ---------------------------------------------------------------------------
// CDNA5 async global->LDS copy (ASYNCcnt) with portable fallback
// ---------------------------------------------------------------------------
#if __has_builtin(__builtin_amdgcn_global_load_async_to_lds_b128)
#define USE_ASYNC 1
typedef int v4i __attribute__((vector_size(16)));
typedef __attribute__((address_space(1))) v4i g_v4i;   // global source
typedef __attribute__((address_space(3))) v4i l_v4i;   // LDS destination
#else
#define USE_ASYNC 0
#endif

__device__ __forceinline__ void async_copy16(const bf16* g, bf16* l) {
#if USE_ASYNC
  __builtin_amdgcn_global_load_async_to_lds_b128(
      (g_v4i*)(void*)g, (l_v4i*)(void*)l, 0, 0);
#else
  *(uint4*)l = *(const uint4*)g;
#endif
}

template <int N>
__device__ __forceinline__ void wait_async() {
#if USE_ASYNC
#if __has_builtin(__builtin_amdgcn_s_wait_asynccnt)
  __builtin_amdgcn_s_wait_asynccnt(N);
#else
  asm volatile("s_wait_asynccnt %0" :: "i"(N) : "memory");
#endif
#endif
}

// ---------------------------------------------------------------------------
// bf16 WMMA GEMM, double-buffered: out = A(bf16,[M,K]) @ W(f32,[K,N]) + bias
// mode 0: outF = acc+bias
// mode 1: outB = bf16(acc+bias)
// mode 2: outF = acc+bias+res
// mode 3: outB = bf16(gelu_tanh(acc+bias))
// mode 4: outF += scale[m*scaleStride] * (acc+bias)
// ---------------------------------------------------------------------------
__global__ __launch_bounds__(256) void gemm_bf16_wmma(
    const bf16* __restrict__ A, const float* __restrict__ W,
    const float* __restrict__ bias,
    float* __restrict__ outF, bf16* __restrict__ outB,
    const float* __restrict__ res,
    const float* __restrict__ scale, int scaleStride,
    int M, int N, int K, int mode)
{
  __shared__ __align__(16) bf16 As[2][BM][KBc + PAD];   // [buf][m][k]
  __shared__ __align__(16) bf16 Bs[2][BN][KBc + PAD];   // [buf][n][k] (transposed)
  const int tid  = threadIdx.x;
  const int lane = tid & 31, wid = tid >> 5;
  const int g = lane >> 4, ln = lane & 15;
  const int wm = wid >> 1, wn = wid & 1;                // 4 x 2 wave grid, 32x64/wave
  const int m0 = blockIdx.y * BM, n0 = blockIdx.x * BN;

  v8f acc[2][4];
#pragma unroll
  for (int ms = 0; ms < 2; ++ms)
#pragma unroll
    for (int ns = 0; ns < 4; ++ns)
#pragma unroll
      for (int i = 0; i < 8; ++i) acc[ms][ns][i] = 0.f;

  const int arow  = tid >> 1, ahalf = (tid & 1) * 16;   // A: 16 bf16/thread (2 x b128)
  const int bkr   = tid >> 3, bnb   = (tid & 7) * 16;   // B: 16 f32/thread

  // ---- prologue: stage tile 0 ----
  {
    const bf16* asrc = A + (size_t)(m0 + arow) * K + ahalf;
    async_copy16(asrc,     &As[0][arow][ahalf]);
    async_copy16(asrc + 8, &As[0][arow][ahalf + 8]);
    const float* bsrc = W + (size_t)bkr * N + n0 + bnb;
#pragma unroll
    for (int j = 0; j < 16; ++j) Bs[0][bnb + j][bkr] = (bf16)bsrc[j];
  }

  const int iters = K / KBc;
  for (int it = 0; it < iters; ++it) {
    const int k0  = it * KBc;
    const int cur = it & 1, nxt = cur ^ 1;
    const bool hasNext = (it + 1) < iters;

    __syncthreads();                       // close previous iteration's LDS reads
    float breg[16];
    if (hasNext) {
      const bf16* asrc = A + (size_t)(m0 + arow) * K + (k0 + KBc) + ahalf;
      async_copy16(asrc,     &As[nxt][arow][ahalf]);      // overlaps with compute
      async_copy16(asrc + 8, &As[nxt][arow][ahalf + 8]);
      const float* bsrc = W + (size_t)(k0 + KBc + bkr) * N + n0 + bnb;
#pragma unroll
      for (int j = 0; j < 16; ++j) breg[j] = bsrc[j];
      if (it + 2 < iters) {
        __builtin_prefetch(W + (size_t)(k0 + 2 * KBc + bkr) * N + n0 + bnb, 0, 1);
        __builtin_prefetch(A + (size_t)(m0 + arow) * K + (k0 + 2 * KBc) + ahalf, 0, 1);
      }
    }
    if (hasNext) wait_async<1>(); else wait_async<0>();   // current buffer's A done
    __syncthreads();                                      // whole tile visible

    v16bf af[2], bfr[4];
#pragma unroll
    for (int ms = 0; ms < 2; ++ms) {
      const int row = wm * 32 + ms * 16 + ln;             // A layout: lane%16 = M row
#pragma unroll
      for (int i = 0; i < 8; ++i) {
        af[ms][i]     = As[cur][row][g * 16 + i];         // K 0-7 / 16-23 by lane half
        af[ms][8 + i] = As[cur][row][g * 16 + 8 + i];     // K 8-15 / 24-31
      }
    }
#pragma unroll
    for (int ns = 0; ns < 4; ++ns) {
      const int col = wn * 64 + ns * 16 + ln;             // B layout: lane%16 = N col
#pragma unroll
      for (int i = 0; i < 16; ++i) bfr[ns][i] = Bs[cur][col][g * 16 + i];
    }
#pragma unroll
    for (int ms = 0; ms < 2; ++ms)
#pragma unroll
      for (int ns = 0; ns < 4; ++ns)
        acc[ms][ns] = __builtin_amdgcn_wmma_f32_16x16x32_bf16(
            false, af[ms], false, bfr[ns], (short)0, acc[ms][ns], false, false);

    if (hasNext) {
#pragma unroll
      for (int j = 0; j < 16; ++j) Bs[nxt][bnb + j][bkr] = (bf16)breg[j];
    }
  }

#pragma unroll
  for (int ms = 0; ms < 2; ++ms)
#pragma unroll
    for (int ns = 0; ns < 4; ++ns)
#pragma unroll
      for (int r = 0; r < 8; ++r) {
        const int mrow = m0 + wm * 32 + ms * 16 + r + 8 * g;  // C/D: M = r (+8 hi half)
        const int ncol = n0 + wn * 64 + ns * 16 + ln;
        const size_t idx = (size_t)mrow * N + ncol;
        float v = acc[ms][ns][r] + bias[ncol];
        if (mode == 0) {
          outF[idx] = v;
        } else if (mode == 1) {
          outB[idx] = (bf16)v;
        } else if (mode == 2) {
          outF[idx] = v + res[idx];
        } else if (mode == 3) {
          // 0.5v(1+tanh(u)) == v*sigmoid(2u), u = 0.7978845608(v + 0.044715 v^3)
          const float u2 = 1.5957691216f * (v + 0.044715f * v * v * v);
          outB[idx] = (bf16)(v / (1.f + __expf(-u2)));
        } else {
          outF[idx] += scale[(size_t)mrow * scaleStride] * v;
        }
      }
}

// ---------------------------------------------------------------------------
// Flash attention, one wave per 16-query tile per (b, h). bf16 Q/K/V in,
// bf16 O out. Scores and PV both via v_wmma_f32_16x16x32_bf16.
// ---------------------------------------------------------------------------
__global__ __launch_bounds__(32) void attn_flash_wmma(
    const bf16* __restrict__ Qb, const bf16* __restrict__ Kb,
    const bf16* __restrict__ Vb, bf16* __restrict__ Ob,
    int Tq, int Tk, int ldq, int ldkv, int Cout, int causal)
{
  const int qtile = blockIdx.x, h = blockIdx.y, b = blockIdx.z;
  const int lane = threadIdx.x, g = lane >> 4, ln = lane & 15;
  const int q0 = qtile * 16;

  const bf16* Qp = Qb + (size_t)b * Tq * ldq + (size_t)h * HDd;
  const bf16* Kp = Kb + (size_t)b * Tk * ldkv + (size_t)h * HDd;
  const bf16* Vp = Vb + (size_t)b * Tk * ldkv + (size_t)h * HDd;

  __shared__ __align__(16) bf16 Vt[HDd][32 + PAD];   // transposed V chunk: [hd][key]
  __shared__ __align__(16) bf16 Pl[16][32 + PAD];    // probabilities: [query][key]

  // Q A-fragments (two 16x32 frags covering HD=64), loaded once
  v16bf qa[2];
  {
    const bf16* src = Qp + (size_t)(q0 + ln) * ldq;
#pragma unroll
    for (int f = 0; f < 2; ++f)
#pragma unroll
      for (int i = 0; i < 8; ++i) {
        qa[f][i]     = src[f * 32 + g * 16 + i];
        qa[f][8 + i] = src[f * 32 + g * 16 + 8 + i];
      }
  }

  float rmax[8], rsum[8];
#pragma unroll
  for (int r = 0; r < 8; ++r) { rmax[r] = -1e30f; rsum[r] = 0.f; }
  v8f accO[4];
#pragma unroll
  for (int j = 0; j < 4; ++j)
#pragma unroll
    for (int i = 0; i < 8; ++i) accO[j][i] = 0.f;

  const int kend = causal ? (q0 + 16) : Tk;
  for (int kc = 0; kc < kend; kc += 32) {
    __syncthreads();
    {  // stage V chunk transposed into LDS
      int srow = kc + lane; if (srow > Tk - 1) srow = Tk - 1;
      const bf16* vsrc = Vp + (size_t)srow * ldkv;
#pragma unroll
      for (int c = 0; c < HDd; ++c) Vt[c][lane] = vsrc[c];
    }
    // two 16-key score subtiles
    float pv[2][8];
#pragma unroll
    for (int st = 0; st < 2; ++st) {
      v8f s;
#pragma unroll
      for (int i = 0; i < 8; ++i) s[i] = 0.f;
      const int key = kc + st * 16 + ln;
      int keyc = key; if (keyc > Tk - 1) keyc = Tk - 1;
      const bf16* ksrc = Kp + (size_t)keyc * ldkv;
      v16bf kf0, kf1;   // B = K^T: b[i] = K[key][f*32 + g*16 + i]
#pragma unroll
      for (int i = 0; i < 16; ++i) {
        kf0[i] = ksrc[g * 16 + i];
        kf1[i] = ksrc[32 + g * 16 + i];
      }
      s = __builtin_amdgcn_wmma_f32_16x16x32_bf16(false, qa[0], false, kf0, (short)0, s, false, false);
      s = __builtin_amdgcn_wmma_f32_16x16x32_bf16(false, qa[1], false, kf1, (short)0, s, false, false);
#pragma unroll
      for (int r = 0; r < 8; ++r) {
        float val = s[r] * 0.125f;                 // 1/sqrt(64)
        const int qi = q0 + r + 8 * g;
        if ((causal && key > qi) || key >= Tk) val = -1e30f;
        pv[st][r] = val;
      }
    }
    // online softmax (rows live in 16-lane halves of the wave)
#pragma unroll
    for (int r = 0; r < 8; ++r) {
      float mx = fmaxf(pv[0][r], pv[1][r]);
#pragma unroll
      for (int off = 8; off > 0; off >>= 1) mx = fmaxf(mx, __shfl_xor(mx, off, 32));
      const float nm  = fmaxf(rmax[r], mx);
      const float scl = __expf(rmax[r] - nm);
      rmax[r] = nm;
      const float p0 = __expf(pv[0][r] - nm);
      const float p1 = __expf(pv[1][r] - nm);
      float rs = p0 + p1;
#pragma unroll
      for (int off = 8; off > 0; off >>= 1) rs += __shfl_xor(rs, off, 32);
      rsum[r] = rsum[r] * scl + rs;
#pragma unroll
      for (int j = 0; j < 4; ++j) accO[j][r] *= scl;
      const int m = r + 8 * g;
      Pl[m][ln]      = (bf16)p0;
      Pl[m][16 + ln] = (bf16)p1;
    }
    __syncthreads();
    // O += P[16x32] @ V[32x64]
    v16bf pf;
#pragma unroll
    for (int i = 0; i < 8; ++i) {
      pf[i]     = Pl[ln][g * 16 + i];
      pf[8 + i] = Pl[ln][g * 16 + 8 + i];
    }
#pragma unroll
    for (int j = 0; j < 4; ++j) {
      v16bf vf;
#pragma unroll
      for (int i = 0; i < 16; ++i) vf[i] = Vt[j * 16 + ln][g * 16 + i];
      accO[j] = __builtin_amdgcn_wmma_f32_16x16x32_bf16(false, pf, false, vf, (short)0, accO[j], false, false);
    }
  }
  // finalize: divide by row sums, write bf16 O
#pragma unroll
  for (int r = 0; r < 8; ++r) {
    const float inv = 1.f / rsum[r];
    const int t = q0 + r + 8 * g;
    bf16* dst = Ob + (size_t)(b * Tq + t) * Cout + (size_t)h * HDd;
#pragma unroll
    for (int j = 0; j < 4; ++j) dst[j * 16 + ln] = (bf16)(accO[j][r] * inv);
  }
}

// ---------------------------------------------------------------------------
// LayerNorm (f32 in, bf16 out), one block per token
// ---------------------------------------------------------------------------
__global__ __launch_bounds__(256) void ln_to_bf16(
    const float* __restrict__ x, const float* __restrict__ gam,
    const float* __restrict__ bet, bf16* __restrict__ out, int Cdim)
{
  const int tok = blockIdx.x, tid = threadIdx.x;
  const float* xp = x + (size_t)tok * Cdim;
  __shared__ float red[256];
  float s = 0.f;
  for (int c = tid; c < Cdim; c += 256) s += xp[c];
  red[tid] = s; __syncthreads();
  for (int off = 128; off > 0; off >>= 1) { if (tid < off) red[tid] += red[tid + off]; __syncthreads(); }
  const float mean = red[0] / Cdim;
  __syncthreads();
  float v = 0.f;
  for (int c = tid; c < Cdim; c += 256) { float d = xp[c] - mean; v += d * d; }
  red[tid] = v; __syncthreads();
  for (int off = 128; off > 0; off >>= 1) { if (tid < off) red[tid] += red[tid + off]; __syncthreads(); }
  const float inv = rsqrtf(red[0] / Cdim + EPSc);
  bf16* op = out + (size_t)tok * Cdim;
  for (int c = tid; c < Cdim; c += 256)
    op[c] = (bf16)((xp[c] - mean) * inv * gam[c] + bet[c]);
}

// ---------------------------------------------------------------------------
// MoE gate: sigmoid scores then top-2 selection (unnormalized weights)
// ---------------------------------------------------------------------------
__global__ __launch_bounds__(256) void gate_scores(
    const bf16* __restrict__ h3, const float* __restrict__ gw,
    const float* __restrict__ gb, float* __restrict__ comb, int Nn)
{
  const int idx = blockIdx.x * 256 + threadIdx.x;
  if (idx >= Nn * Ee) return;
  const int n = idx / Ee, e = idx % Ee;
  float s = gb[e];
  const bf16* xp = h3 + (size_t)n * Cc;
  for (int c = 0; c < Cc; ++c) s += (float)xp[c] * gw[(size_t)c * Ee + e];
  comb[idx] = 1.f / (1.f + __expf(-s));
}

__global__ __launch_bounds__(256) void gate_top2(float* __restrict__ comb, int Nn)
{
  const int n = blockIdx.x * 256 + threadIdx.x;
  if (n >= Nn) return;
  float sc[Ee];
#pragma unroll
  for (int e = 0; e < Ee; ++e) sc[e] = comb[(size_t)n * Ee + e];
  int i1 = 0;
#pragma unroll
  for (int e = 1; e < Ee; ++e) if (sc[e] > sc[i1]) i1 = e;
  int i2 = -1;
#pragma unroll
  for (int e = 0; e < Ee; ++e) {
    if (e == i1) continue;
    if (i2 < 0 || sc[e] > sc[i2]) i2 = e;
  }
#pragma unroll
  for (int e = 0; e < Ee; ++e)
    comb[(size_t)n * Ee + e] = (e == i1 || e == i2) ? sc[e] : 0.f;
}

// ---------------------------------------------------------------------------
extern "C" void kernel_launch(void* const* d_in, const int* in_sizes, int n_in,
                              void* d_out, int out_size, void* d_ws, size_t ws_size,
                              hipStream_t stream)
{
  (void)in_sizes; (void)n_in; (void)out_size; (void)ws_size;
  const float* x       = (const float*)d_in[0];
  const float* imgs    = (const float*)d_in[1];
  /* d_in[2] = dis_logits, unused by reference */
  const float* ln1_g   = (const float*)d_in[3];  const float* ln1_b = (const float*)d_in[4];
  const float* ln2_g   = (const float*)d_in[5];  const float* ln2_b = (const float*)d_in[6];
  const float* ln3_g   = (const float*)d_in[7];  const float* ln3_b = (const float*)d_in[8];
  const float* attn_w  = (const float*)d_in[9];  const float* attn_b  = (const float*)d_in[10];
  const float* attn_pw = (const float*)d_in[11]; const float* attn_pb = (const float*)d_in[12];
  const float* kv_w    = (const float*)d_in[13]; const float* kv_b    = (const float*)d_in[14];
  const float* q_w     = (const float*)d_in[15]; const float* q_b     = (const float*)d_in[16];
  const float* cp_w    = (const float*)d_in[17]; const float* cp_b    = (const float*)d_in[18];
  const float* gate_w  = (const float*)d_in[19]; const float* gate_b  = (const float*)d_in[20];
  const float* w1      = (const float*)d_in[21]; const float* b1      = (const float*)d_in[22];
  const float* w2      = (const float*)d_in[23]; const float* b2      = (const float*)d_in[24];
  float* out = (float*)d_out;

  char* wsp = (char*)d_ws;
  auto alloc = [&](size_t bytes) -> void* {
    void* p = (void*)wsp; wsp += (bytes + 255) & ~(size_t)255; return p;
  };
  bf16*  h1   = (bf16*) alloc((size_t)Ntok * Cc * 2);        // ln1(x)
  bf16*  qkv  = (bf16*) alloc((size_t)Ntok * 3 * Cc * 2);    // q|k|v
  bf16*  aO   = (bf16*) alloc((size_t)Ntok * Cc * 2);        // self-attn out
  float* x1   = (float*)alloc((size_t)Ntok * Cc * 4);        // x + attn
  bf16*  hi   = (bf16*) alloc((size_t)NimgT * Cc * 2);       // ln2(imgs)
  bf16*  kvb  = (bf16*) alloc((size_t)NimgT * 2 * Cc * 2);   // k|v cross
  bf16*  h2   = (bf16*) alloc((size_t)Ntok * Cc * 2);        // ln2(x1)
  bf16*  qb   = (bf16*) alloc((size_t)Ntok * Cc * 2);        // q cross
  bf16*  cO   = (bf16*) alloc((size_t)Ntok * Cc * 2);        // cross-attn out
  bf16*  h3   = (bf16*) alloc((size_t)Ntok * Cc * 2);        // ln3(x2)
  float* comb = (float*)alloc((size_t)Ntok * Ee * 4);        // gate weights
  bf16*  hid  = (bf16*) alloc((size_t)Ntok * FFf * 2);       // expert hidden

  dim3 blk256(256), blk32(32);

  // --- causal self-attention ---
  ln_to_bf16<<<Ntok, blk256, 0, stream>>>(x, ln1_g, ln1_b, h1, Cc);
  gemm_bf16_wmma<<<dim3(3 * Cc / BN, Ntok / BM), blk256, 0, stream>>>(
      h1, attn_w, attn_b, nullptr, qkv, nullptr, nullptr, 0, Ntok, 3 * Cc, Cc, 1);
  attn_flash_wmma<<<dim3(Tt / 16, Hh, Bq), blk32, 0, stream>>>(
      qkv, qkv + Cc, qkv + 2 * Cc, aO, Tt, Tt, 3 * Cc, 3 * Cc, Cc, 1);
  gemm_bf16_wmma<<<dim3(Cc / BN, Ntok / BM), blk256, 0, stream>>>(
      aO, attn_pw, attn_pb, x1, nullptr, x, nullptr, 0, Ntok, Cc, Cc, 2);

  // --- cross-attention ---
  ln_to_bf16<<<NimgT, blk256, 0, stream>>>(imgs, ln2_g, ln2_b, hi, Cc);
  gemm_bf16_wmma<<<dim3(2 * Cc / BN, NimgT / BM), blk256, 0, stream>>>(
      hi, kv_w, kv_b, nullptr, kvb, nullptr, nullptr, 0, NimgT, 2 * Cc, Cc, 1);
  ln_to_bf16<<<Ntok, blk256, 0, stream>>>(x1, ln2_g, ln2_b, h2, Cc);
  gemm_bf16_wmma<<<dim3(Cc / BN, Ntok / BM), blk256, 0, stream>>>(
      h2, q_w, q_b, nullptr, qb, nullptr, nullptr, 0, Ntok, Cc, Cc, 1);
  attn_flash_wmma<<<dim3(Tt / 16, Hh, Bq), blk32, 0, stream>>>(
      qb, kvb, kvb + Cc, cO, Tt, TIMGc, Cc, 2 * Cc, Cc, 0);
  gemm_bf16_wmma<<<dim3(Cc / BN, Ntok / BM), blk256, 0, stream>>>(
      cO, cp_w, cp_b, out, nullptr, x1, nullptr, 0, Ntok, Cc, Cc, 2);   // out = x2

  // --- MoE ---
  ln_to_bf16<<<Ntok, blk256, 0, stream>>>(out, ln3_g, ln3_b, h3, Cc);
  gate_scores<<<(Ntok * Ee + 255) / 256, blk256, 0, stream>>>(h3, gate_w, gate_b, comb, Ntok);
  gate_top2<<<(Ntok + 255) / 256, blk256, 0, stream>>>(comb, Ntok);
  for (int e = 0; e < Ee; ++e) {
    gemm_bf16_wmma<<<dim3(FFf / BN, Ntok / BM), blk256, 0, stream>>>(
        h3, w1 + (size_t)e * Cc * FFf, b1 + (size_t)e * FFf,
        nullptr, hid, nullptr, nullptr, 0, Ntok, FFf, Cc, 3);
    gemm_bf16_wmma<<<dim3(Cc / BN, Ntok / BM), blk256, 0, stream>>>(
        hid, w2 + (size_t)e * FFf * Cc, b2 + (size_t)e * Cc,
        out, nullptr, nullptr, comb + e, Ee, Ntok, Cc, FFf, 4);
  }
}
